// GKFormerV0_48722109005875
// MI455X (gfx1250) — matrix operand order repaired and checked by reading, not compiled, not run on hardware
//
#include <hip/hip_runtime.h>
#include <hip/hip_bf16.h>
#include <math.h>
#include <stdint.h>

// ---------------------------------------------------------------------------
// ViT forward for MI455X (gfx1250): WMMA GEMMs + WMMA attention, with
// CDNA5 async global->LDS staging (GLOBAL_LOAD_ASYNC_TO_LDS_B128 / ASYNCcnt)
// and double-buffered LDS pipelining.
// ---------------------------------------------------------------------------

typedef _Float16 h16;
typedef _Float16 v16h __attribute__((ext_vector_type(16)));
typedef float    v8f  __attribute__((ext_vector_type(8)));
typedef unsigned int v8u __attribute__((ext_vector_type(8)));

union FragU { v8u u; v16h h; };

#define BN   128
#define L2T  225
#define LT   226         // tokens per image
#define CT   512
#define NHD  8
#define HDD  64
#define DL   12
#define HIDN 2048
#define LP   256         // padded token count for attention (16 WMMA tiles)
#define MROWS (BN*LT)    // 28928 = 226*128

// GEMM macro-tile
#define GMT 128
#define GNT 64
#define GKT 64

// ---------------- WMMA fragment gathers (ISA 7.12.2 layouts) ----------------

// A-matrix 16x32 f16: element (m,k) at p[m*stride + k]
__device__ __forceinline__ v16h frag_a(const h16* p, int stride, int lane) {
  int m = lane & 15, hg = lane >> 4;
  FragU r;
#pragma unroll
  for (int i = 0; i < 8; ++i) {
    int blk = i >> 2, j = i & 3;
    int k = blk * 16 + hg * 8 + 2 * j;
    r.u[i] = *(const unsigned int*)(p + m * stride + k);
  }
  return r.h;
}

// B-matrix 32x16 f16 stored as Bt (n,k): element at p[n*stride + k]
__device__ __forceinline__ v16h frag_b(const h16* p, int stride, int lane) {
  int n = lane & 15, hg = lane >> 4;
  FragU r;
#pragma unroll
  for (int i = 0; i < 8; ++i) {
    int k = (i >> 2) * 16 + hg * 8 + 2 * (i & 3);
    r.u[i] = *(const unsigned int*)(p + n * stride + k);
  }
  return r.h;
}

__device__ __forceinline__ v8f wmma32(v16h a, v16h b, v8f c) {
  return __builtin_amdgcn_wmma_f32_16x16x32_f16(false, a, false, b, (short)0, c,
                                                false, false);
}

__device__ __forceinline__ float gelu_exact(float x) {
  return 0.5f * x * (1.0f + erff(x * 0.70710678118654752f));
}

// --------- CDNA5 async copy: 16B per lane, global -> LDS (ASYNCcnt) ---------
__device__ __forceinline__ void async_copy_b128(const h16* gptr, h16* lptr) {
  unsigned lds = (unsigned)(uintptr_t)lptr;   // low 32 bits = LDS offset
  asm volatile("global_load_async_to_lds_b128 %0, %1, off"
               :: "v"(lds), "v"(gptr) : "memory");
}

// ---------------- weight prep: f32 (K,N) -> f16 (N,K) -----------------------
__global__ void k_transpose_w(const float* __restrict__ src, h16* __restrict__ dst,
                              int K, int N) {
  int idx = blockIdx.x * blockDim.x + threadIdx.x;
  if (idx >= N * K) return;
  int n = idx / K, k = idx - n * K;
  dst[idx] = (h16)src[k * N + n];
}

// ---------------- RoPE cos/sin tables ---------------------------------------
__global__ void k_rope_tables(const float* __restrict__ freqs,
                              const float* __restrict__ cls_freq,
                              float* __restrict__ cosT, float* __restrict__ sinT) {
  int idx = blockIdx.x * blockDim.x + threadIdx.x;
  if (idx >= LT * HDD) return;
  int l = idx / HDD, d = idx - l * HDD;
  float f = (l < L2T) ? freqs[idx] : cls_freq[d];
  cosT[idx] = cosf(f);
  sinT[idx] = sinf(f);
}

// ---------------- embedding --------------------------------------------------
__global__ void k_embed(const float* __restrict__ x, const float* __restrict__ coords,
                        const float* __restrict__ in_w, const float* __restrict__ in_b,
                        const float* __restrict__ cls_tok, float* __restrict__ h) {
  int idx = blockIdx.x * blockDim.x + threadIdx.x;
  if (idx >= MROWS * CT) return;
  int c = idx & (CT - 1);
  int r = idx >> 9;               // row = b*226 + l
  int l = r % LT, b = r / LT;
  if (l == L2T) { h[idx] = cls_tok[c]; return; }
  float x0 = x[(b * 3 + 0) * L2T + l];
  float x1 = x[(b * 3 + 1) * L2T + l];
  float x2 = x[(b * 3 + 2) * L2T + l];
  float cy = coords[l * 2 + 0], cx = coords[l * 2 + 1];
  h[idx] = x0 * in_w[0 * CT + c] + x1 * in_w[1 * CT + c] + x2 * in_w[2 * CT + c] +
           cy * in_w[3 * CT + c] + cx * in_w[4 * CT + c] + in_b[c];
}

// ---------------- layernorm: one wave per 512-wide row, out f16 -------------
__global__ void k_ln(const float* __restrict__ h, const float* __restrict__ g,
                     const float* __restrict__ bta, h16* __restrict__ out, int M) {
  int gw = (blockIdx.x * blockDim.x + threadIdx.x) >> 5;
  int lane = threadIdx.x & 31;
  if (gw >= M) return;
  const float* row = h + (size_t)gw * CT;
  float v[16];
  float s = 0.f;
#pragma unroll
  for (int i = 0; i < 16; ++i) { v[i] = row[lane + 32 * i]; s += v[i]; }
#pragma unroll
  for (int off = 16; off >= 1; off >>= 1) s += __shfl_xor(s, off, 32);
  float mean = s * (1.0f / CT);
  float ss = 0.f;
#pragma unroll
  for (int i = 0; i < 16; ++i) { float d = v[i] - mean; ss += d * d; }
#pragma unroll
  for (int off = 16; off >= 1; off >>= 1) ss += __shfl_xor(ss, off, 32);
  float inv = rsqrtf(ss * (1.0f / CT) + 1e-5f);
#pragma unroll
  for (int i = 0; i < 16; ++i) {
    int c = lane + 32 * i;
    out[(size_t)gw * CT + c] = (h16)((v[i] - mean) * inv * g[c] + bta[c]);
  }
}

// ---------------- f32 -> f16 convert ----------------------------------------
__global__ void k_f32_to_f16(const float* __restrict__ src, h16* __restrict__ dst,
                             int n) {
  int idx = blockIdx.x * blockDim.x + threadIdx.x;
  if (idx < n) dst[idx] = (h16)src[idx];
}

// ---------------- WMMA GEMM: A(M,K)f16 @ Bt(N,K)f16 + bias ------------------
// 128x64 macro-tile per block (8 waves, each 32x32 = 4 accumulators), K-tile 64.
// Double-buffered LDS fed by global_load_async_to_lds_b128 (ASYNCcnt pipeline).
// mode 0: store f16     mode 1: store f16(gelu)     mode 2: store f32 + resid
__global__ __launch_bounds__(256)
void k_gemm(const h16* __restrict__ A, const h16* __restrict__ Bt,
            const float* __restrict__ bias, void* __restrict__ out,
            const float* __restrict__ resid, int M, int N, int K, int mode) {
  __shared__ __align__(16) h16 lA[2][GMT * GKT];  // 2 x 16 KB
  __shared__ __align__(16) h16 lB[2][GNT * GKT];  // 2 x 8 KB
  int tid = threadIdx.x;
  int bm = blockIdx.x, bn = blockIdx.y;
  int wv = tid >> 5, lane = tid & 31;
  int mw = wv & 3, nw = wv >> 2;                  // 32-row / 32-col wave region

  const h16* Ab = A + (size_t)bm * GMT * K;       // + row*K + kb*64 + col
  const h16* Bb = Bt + (size_t)bn * GNT * K;

  // per-lane staging coordinates (16B units)
  int fa[4], fb[2];
#pragma unroll
  for (int i = 0; i < 4; ++i) fa[i] = i * 256 + tid;   // A: 1024 units
#pragma unroll
  for (int i = 0; i < 2; ++i) fb[i] = i * 256 + tid;   // B: 512 units

  auto stage = [&](int kb, int buf) {
#pragma unroll
    for (int i = 0; i < 4; ++i) {
      int row = fa[i] >> 3, col = (fa[i] & 7) * 8;
      async_copy_b128(Ab + (size_t)row * K + kb * GKT + col,
                      &lA[buf][row * GKT + col]);
    }
#pragma unroll
    for (int i = 0; i < 2; ++i) {
      int row = fb[i] >> 3, col = (fb[i] & 7) * 8;
      async_copy_b128(Bb + (size_t)row * K + kb * GKT + col,
                      &lB[buf][row * GKT + col]);
    }
  };

  v8f acc00 = {}, acc01 = {}, acc10 = {}, acc11 = {};
  int ksteps = K >> 6;                            // K/64
  stage(0, 0);                                    // prologue: tile 0 in flight
  for (int kb = 0; kb < ksteps; ++kb) {
    int cur = kb & 1;
    if (kb + 1 < ksteps) {
      stage(kb + 1, cur ^ 1);                     // next tile in flight
      asm volatile("s_wait_asynccnt 6" ::: "memory");   // current tile resident
    } else {
      asm volatile("s_wait_asynccnt 0" ::: "memory");
    }
    __syncthreads();                              // all waves' data visible
#pragma unroll
    for (int ks = 0; ks < GKT; ks += 32) {
      v16h a0 = frag_a(&lA[cur][(mw * 32 + 0) * GKT + ks], GKT, lane);
      v16h a1 = frag_a(&lA[cur][(mw * 32 + 16) * GKT + ks], GKT, lane);
      v16h b0 = frag_b(&lB[cur][(nw * 32 + 0) * GKT + ks], GKT, lane);
      v16h b1 = frag_b(&lB[cur][(nw * 32 + 16) * GKT + ks], GKT, lane);
      acc00 = wmma32(a0, b0, acc00);
      acc01 = wmma32(a0, b1, acc01);
      acc10 = wmma32(a1, b0, acc10);
      acc11 = wmma32(a1, b1, acc11);
    }
    __syncthreads();                              // protect buffer reuse
  }

  int n = lane & 15, hg = lane >> 4;
  v8f accs[2][2] = {{acc00, acc01}, {acc10, acc11}};
#pragma unroll
  for (int ti = 0; ti < 2; ++ti) {
#pragma unroll
    for (int tj = 0; tj < 2; ++tj) {
      int gcol = bn * GNT + nw * 32 + tj * 16 + n;
      float bv = bias ? bias[gcol] : 0.f;
#pragma unroll
      for (int i = 0; i < 8; ++i) {
        int grow = bm * GMT + mw * 32 + ti * 16 + hg * 8 + i;
        size_t idx = (size_t)grow * N + gcol;
        float val = accs[ti][tj][i] + bv;
        if (mode == 0) {
          ((h16*)out)[idx] = (h16)val;
        } else if (mode == 1) {
          ((h16*)out)[idx] = (h16)gelu_exact(val);
        } else {
          ((float*)out)[idx] = val + resid[idx];
        }
      }
    }
  }
}

// ---------------- qkv (M,1536)f16 -> rope'd Q,K (bh,LP,64) + Vt (bh,64,LP) --
__global__ void k_rope_qkv(const h16* __restrict__ qkv,
                           const float* __restrict__ cosT,
                           const float* __restrict__ sinT,
                           h16* __restrict__ Qb, h16* __restrict__ Kb,
                           h16* __restrict__ Vtb) {
  int idx = blockIdx.x * blockDim.x + threadIdx.x;       // B*NH*LP*32
  int hd2 = idx & 31;
  int lp = (idx >> 5) & (LP - 1);
  int nh = (idx >> 13) & (NHD - 1);
  int b = idx >> 16;
  if (b >= BN) return;
  int hd = hd2 * 2;
  int bh = b * NHD + nh;
  h16* Qp = Qb + ((size_t)bh * LP + lp) * HDD + hd;
  h16* Kp = Kb + ((size_t)bh * LP + lp) * HDD + hd;
  h16* Vp = Vtb + (size_t)bh * HDD * LP + (size_t)hd * LP + lp;
  if (lp >= LT) {
    Qp[0] = (h16)0.f; Qp[1] = (h16)0.f;
    Kp[0] = (h16)0.f; Kp[1] = (h16)0.f;
    Vp[0] = (h16)0.f; Vp[LP] = (h16)0.f;
    return;
  }
  const h16* src = qkv + (size_t)(b * LT + lp) * (3 * CT);
  int col = nh * HDD + hd;
  float q0 = (float)src[col],            q1 = (float)src[col + 1];
  float k0 = (float)src[CT + col],       k1 = (float)src[CT + col + 1];
  float v0 = (float)src[2 * CT + col],   v1 = (float)src[2 * CT + col + 1];
  float c0 = cosT[lp * HDD + hd], c1 = cosT[lp * HDD + hd + 1];
  float s0 = sinT[lp * HDD + hd], s1 = sinT[lp * HDD + hd + 1];
  Qp[0] = (h16)(q0 * c0 + q1 * s0);
  Qp[1] = (h16)(q1 * c1 - q0 * s1);
  Kp[0] = (h16)(k0 * c0 + k1 * s0);
  Kp[1] = (h16)(k1 * c1 - k0 * s1);
  Vp[0] = (h16)v0;
  Vp[LP] = (h16)v1;
}

// ---------------- attention: one wave = one 16-query tile -------------------
__global__ __launch_bounds__(128)
void k_attn(const h16* __restrict__ Qb, const h16* __restrict__ Kb,
            const h16* __restrict__ Vtb, h16* __restrict__ Ob) {
  __shared__ __align__(16) h16 P[4][16 * LP];   // 32 KB
  int bh = blockIdx.x;
  int b = bh >> 3, nh = bh & 7;
  int wv = threadIdx.x >> 5, lane = threadIdx.x & 31;
  int qt = blockIdx.y * 4 + wv;
  int q0 = qt * 16;
  int nlo = lane & 15, hg = lane >> 4;

  const h16* Qp = Qb + ((size_t)bh * LP + q0) * HDD;
  v16h qa0 = frag_a(Qp, HDD, lane);
  v16h qa1 = frag_a(Qp + 32, HDD, lane);

  float sc[16][8];
#pragma unroll
  for (int t = 0; t < 16; ++t) {
    const h16* Kp = Kb + ((size_t)bh * LP + t * 16) * HDD;
    v8f acc = {};
    acc = wmma32(qa0, frag_b(Kp, HDD, lane), acc);
    acc = wmma32(qa1, frag_b(Kp + 32, HDD, lane), acc);
    bool pad = (t * 16 + nlo) >= LT;      // key column padding
#pragma unroll
    for (int i = 0; i < 8; ++i) sc[t][i] = pad ? -1e30f : acc[i] * 0.125f;
  }

  float rmax[8], rsum[8];
#pragma unroll
  for (int i = 0; i < 8; ++i) {
    float m = sc[0][i];
#pragma unroll
    for (int t = 1; t < 16; ++t) m = fmaxf(m, sc[t][i]);
    rmax[i] = m;
  }
#pragma unroll
  for (int off = 8; off >= 1; off >>= 1)
#pragma unroll
    for (int i = 0; i < 8; ++i) rmax[i] = fmaxf(rmax[i], __shfl_xor(rmax[i], off, 32));
#pragma unroll
  for (int i = 0; i < 8; ++i) {
    float s = 0.f;
#pragma unroll
    for (int t = 0; t < 16; ++t) { sc[t][i] = __expf(sc[t][i] - rmax[i]); s += sc[t][i]; }
    rsum[i] = s;
  }
#pragma unroll
  for (int off = 8; off >= 1; off >>= 1)
#pragma unroll
    for (int i = 0; i < 8; ++i) rsum[i] += __shfl_xor(rsum[i], off, 32);

#pragma unroll
  for (int i = 0; i < 8; ++i) {
    float inv = 1.0f / rsum[i];
    int m = hg * 8 + i;
#pragma unroll
    for (int t = 0; t < 16; ++t)
      P[wv][m * LP + t * 16 + nlo] = (h16)(sc[t][i] * inv);
  }
  __syncthreads();

  const h16* Vp = Vtb + (size_t)bh * HDD * LP;
#pragma unroll
  for (int nt = 0; nt < 4; ++nt) {
    v8f acc = {};
#pragma unroll
    for (int ks = 0; ks < 8; ++ks) {
      v16h a = frag_a(&P[wv][ks * 32], LP, lane);
      v16h bb = frag_b(Vp + (size_t)(nt * 16) * LP + ks * 32, LP, lane);
      acc = wmma32(a, bb, acc);
    }
#pragma unroll
    for (int i = 0; i < 8; ++i) {
      int q = q0 + hg * 8 + i;
      if (q < LT)
        Ob[((size_t)(b * LT + q)) * CT + nh * HDD + nt * 16 + nlo] = (h16)acc[i];
    }
  }
}

// ---------------- action head final reduce: (M,512)f16 . a2_w ---------------
__global__ void k_a2(const h16* __restrict__ g16, const float* __restrict__ a2w,
                     const float* __restrict__ a2b, float* __restrict__ out) {
  int r = (blockIdx.x * blockDim.x + threadIdx.x) >> 5;
  int lane = threadIdx.x & 31;
  if (r >= MROWS) return;
  const h16* row = g16 + (size_t)r * CT;
  float s = 0.f;
#pragma unroll
  for (int i = 0; i < 16; ++i) {
    int c = lane + 32 * i;
    s += (float)row[c] * a2w[c];
  }
#pragma unroll
  for (int off = 16; off >= 1; off >>= 1) s += __shfl_xor(s, off, 32);
  int l = r % LT, b = r / LT;
  if (lane == 0 && l < L2T) out[b * L2T + l] = s + a2b[0];
}

// ---------------- value head: per-batch block, cls token only ---------------
__global__ __launch_bounds__(256)
void k_vhead(const float* __restrict__ h, const float* __restrict__ v1w,
             const float* __restrict__ v1b, const float* __restrict__ v2w,
             const float* __restrict__ v2b, float* __restrict__ out) {
  __shared__ float cls[CT];
  __shared__ float red[256];
  int b = blockIdx.x, tid = threadIdx.x;
  const float* src = h + (size_t)(b * LT + L2T) * CT;
  cls[tid] = src[tid];
  cls[tid + 256] = src[tid + 256];
  __syncthreads();
  float part = 0.f;
  for (int jj = tid; jj < CT; jj += 256) {
    float a = v1b[jj];
    for (int i = 0; i < CT; ++i) a += cls[i] * v1w[i * CT + jj];
    part += gelu_exact(a) * v2w[jj];
  }
  red[tid] = part;
  __syncthreads();
  for (int off = 128; off >= 1; off >>= 1) {
    if (tid < off) red[tid] += red[tid + off];
    __syncthreads();
  }
  if (tid == 0) out[BN * L2T + b] = red[0] + v2b[0];
}

// ---------------------------------------------------------------------------
extern "C" void kernel_launch(void* const* d_in, const int* in_sizes, int n_in,
                              void* d_out, int out_size, void* d_ws, size_t ws_size,
                              hipStream_t stream) {
  const float* x        = (const float*)d_in[0];
  const float* coords   = (const float*)d_in[1];
  const float* freqs    = (const float*)d_in[2];
  const float* in_w     = (const float*)d_in[3];
  const float* in_b     = (const float*)d_in[4];
  const float* cls_tok  = (const float*)d_in[5];
  const float* cls_freq = (const float*)d_in[6];
  const float* ln1_g    = (const float*)d_in[7];
  const float* ln1_b    = (const float*)d_in[8];
  const float* qkv_w    = (const float*)d_in[9];
  const float* qkv_b    = (const float*)d_in[10];
  const float* proj_w   = (const float*)d_in[11];
  const float* proj_b   = (const float*)d_in[12];
  const float* ln2_g    = (const float*)d_in[13];
  const float* ln2_b    = (const float*)d_in[14];
  const float* fc1_w    = (const float*)d_in[15];
  const float* fc1_b    = (const float*)d_in[16];
  const float* fc2_w    = (const float*)d_in[17];
  const float* fc2_b    = (const float*)d_in[18];
  const float* a1_w     = (const float*)d_in[19];
  const float* a1_b     = (const float*)d_in[20];
  const float* a2_w     = (const float*)d_in[21];
  const float* a2_b     = (const float*)d_in[22];
  const float* v1_w     = (const float*)d_in[23];
  const float* v1_b     = (const float*)d_in[24];
  const float* v2_w     = (const float*)d_in[25];
  const float* v2_b     = (const float*)d_in[26];
  float* out = (float*)d_out;

  char* ws = (char*)d_ws;
  size_t off = 0;
  auto alloc = [&](size_t bytes) {
    size_t o = off;
    off = (off + bytes + 255) & ~(size_t)255;
    return o;
  };

  const size_t LSTR = 3145728;  // f16 elems per layer of prepped weights
  h16*   W16  = (h16*)(ws + alloc(DL * LSTR * sizeof(h16)));
  h16*   A1T  = (h16*)(ws + alloc((size_t)CT * CT * sizeof(h16)));
  float* COS  = (float*)(ws + alloc((size_t)LT * HDD * sizeof(float)));
  float* SIN  = (float*)(ws + alloc((size_t)LT * HDD * sizeof(float)));
  float* H    = (float*)(ws + alloc((size_t)MROWS * CT * sizeof(float)));
  h16*   BUFB = (h16*)(ws + alloc((size_t)MROWS * CT * sizeof(h16)));
  h16*   BUFA = (h16*)(ws + alloc((size_t)MROWS * HIDN * sizeof(h16)));
  h16*   QB   = (h16*)(ws + alloc((size_t)BN * NHD * LP * HDD * sizeof(h16)));
  h16*   KB   = (h16*)(ws + alloc((size_t)BN * NHD * LP * HDD * sizeof(h16)));
  h16*   VTB  = (h16*)(ws + alloc((size_t)BN * NHD * HDD * LP * sizeof(h16)));

  // ---- one-time weight prep: transpose f32 (K,N) -> f16 (N,K) ----
  for (int l = 0; l < DL; ++l) {
    k_transpose_w<<<(CT * 3 * CT) / 256, 256, 0, stream>>>(
        qkv_w + (size_t)l * CT * 3 * CT, W16 + (size_t)l * LSTR + 0, CT, 3 * CT);
    k_transpose_w<<<(CT * CT) / 256, 256, 0, stream>>>(
        proj_w + (size_t)l * CT * CT, W16 + (size_t)l * LSTR + 786432, CT, CT);
    k_transpose_w<<<(CT * HIDN) / 256, 256, 0, stream>>>(
        fc1_w + (size_t)l * CT * HIDN, W16 + (size_t)l * LSTR + 1048576, CT, HIDN);
    k_transpose_w<<<(HIDN * CT) / 256, 256, 0, stream>>>(
        fc2_w + (size_t)l * HIDN * CT, W16 + (size_t)l * LSTR + 2097152, HIDN, CT);
  }
  k_transpose_w<<<(CT * CT) / 256, 256, 0, stream>>>(a1_w, A1T, CT, CT);

  k_rope_tables<<<(LT * HDD + 255) / 256, 256, 0, stream>>>(freqs, cls_freq, COS, SIN);
  k_embed<<<(MROWS * CT) / 256, 256, 0, stream>>>(x, coords, in_w, in_b, cls_tok, H);

  const int M = MROWS;                      // 28928 = 226*128
  for (int l = 0; l < DL; ++l) {
    const h16* qkvT  = W16 + (size_t)l * LSTR + 0;
    const h16* projT = W16 + (size_t)l * LSTR + 786432;
    const h16* fc1T  = W16 + (size_t)l * LSTR + 1048576;
    const h16* fc2T  = W16 + (size_t)l * LSTR + 2097152;

    k_ln<<<M / 8, 256, 0, stream>>>(H, ln1_g + l * CT, ln1_b + l * CT, BUFB, M);
    k_gemm<<<dim3(M / GMT, (3 * CT) / GNT), 256, 0, stream>>>(
        BUFB, qkvT, qkv_b + (size_t)l * 3 * CT, BUFA, nullptr, M, 3 * CT, CT, 0);
    k_rope_qkv<<<(BN * NHD * LP * 32) / 256, 256, 0, stream>>>(BUFA, COS, SIN, QB, KB, VTB);
    k_attn<<<dim3(BN * NHD, 4), 128, 0, stream>>>(QB, KB, VTB, BUFB);
    k_gemm<<<dim3(M / GMT, CT / GNT), 256, 0, stream>>>(
        BUFB, projT, proj_b + (size_t)l * CT, H, H, M, CT, CT, 2);
    k_ln<<<M / 8, 256, 0, stream>>>(H, ln2_g + l * CT, ln2_b + l * CT, BUFB, M);
    k_gemm<<<dim3(M / GMT, HIDN / GNT), 256, 0, stream>>>(
        BUFB, fc1T, fc1_b + (size_t)l * HIDN, BUFA, nullptr, M, HIDN, CT, 1);
    k_gemm<<<dim3(M / GMT, CT / GNT), 256, 0, stream>>>(
        BUFA, fc2T, fc2_b + (size_t)l * CT, H, H, M, CT, HIDN, 2);
  }

  // action head: gelu(h @ a1 + b) then . a2
  k_f32_to_f16<<<(M * CT) / 256, 256, 0, stream>>>(H, BUFB, M * CT);
  k_gemm<<<dim3(M / GMT, CT / GNT), 256, 0, stream>>>(
      BUFB, A1T, a1_b, BUFA, nullptr, M, CT, CT, 1);
  k_a2<<<M / 8, 256, 0, stream>>>(BUFA, a2_w, a2_b, out);

  // value head on cls tokens
  k_vhead<<<BN, 256, 0, stream>>>(H, v1_w, v1_b, v2_w, v2_b, out);
}
